// NAttention_9337258901738
// MI455X (gfx1250) — compile-verified, matrix operand order checked by reading, not run on hardware
//
#include <hip/hip_runtime.h>
#include <cstdint>
#include <cstddef>

// ---------- CDNA5 WMMA types ----------
typedef __attribute__((ext_vector_type(16))) __bf16         v16bf;
typedef __attribute__((ext_vector_type(16))) unsigned short v16u;
typedef __attribute__((ext_vector_type(8)))  float          v8f;

#define DIM    256
#define HEADS  8
#define HD     32
#define BATCH  2
#define HH     64
#define WW     64
#define PIX    4096            // H*W
#define NTOK   8192            // B*H*W
#define NCOL   22              // max key-column span for a 16-query row tile
#define NKEY   160             // 7*22=154 keys padded to 10 tiles of 16
#define QSCALE 0.17677669529663687f  // 32^-0.5

// round-to-nearest-even f32 -> bf16 bits
__device__ __forceinline__ unsigned short f2bf(float f) {
  union { float f; unsigned u; } v; v.f = f;
  unsigned u = v.u;
  return (unsigned short)((u + 0x7fffu + ((u >> 16) & 1u)) >> 16);
}

// Build a 16x32(bf16) A/B operand for one lane: two contiguous 8-element runs
// at K offsets 8g and 16+8g (ISA 7.12.2 16-bit A-matrix layout; B symmetric).
__device__ __forceinline__ v16bf pack16(const unsigned short* p0,
                                        const unsigned short* p1) {
  v16u v;
#pragma unroll
  for (int e = 0; e < 8; ++e) { v[e] = p0[e]; v[8 + e] = p1[e]; }
  return __builtin_bit_cast(v16bf, v);
}

__device__ __forceinline__ v8f wmma_bf16(v16bf a, v16bf b, v8f c) {
  return __builtin_amdgcn_wmma_f32_16x16x32_bf16(
      /*neg_a=*/false, a, /*neg_b=*/false, b,
      /*c_mod=*/(short)0, c, /*reuse_a=*/false, /*reuse_b=*/false);
}

// ---------- 1) layout/precision prep: x -> token-major bf16, weights -> bf16
__global__ void prep_kernel(const float* __restrict__ x,
                            const float* __restrict__ wq,
                            const float* __restrict__ wp,
                            unsigned short* __restrict__ xT,
                            unsigned short* __restrict__ wqb,
                            unsigned short* __restrict__ wpb) {
  const int NX = NTOK * DIM;       // 2097152
  const int NW = 3 * DIM * DIM;    // 196608
  int t = blockIdx.x * blockDim.x + threadIdx.x;
  if (t < NX) {
    int token = t >> 8, cc = t & 255;
    int b = token >> 12, pix = token & 4095;
    xT[t] = f2bf(x[((size_t)(b * DIM + cc)) * PIX + pix]);
  } else if (t < NX + NW) {
    int u = t - NX;
    wqb[u] = f2bf(wq[u]);
  } else if (t < NX + NW + DIM * DIM) {
    int u = t - NX - NW;
    wpb[u] = f2bf(wp[u]);
  }
}

// ---------- 2) QKV GEMM: [8192x256]x[256x768]^T, scatter to (b,h,pix,hd) ----
// block = 128 threads = 4 waves in 2x2; wave tile 32x32 (4 accumulators)
__global__ void qkv_gemm(const unsigned short* __restrict__ xT,
                         const unsigned short* __restrict__ wq,
                         const float* __restrict__ bqkv,
                         unsigned short* __restrict__ qb,
                         unsigned short* __restrict__ kb,
                         unsigned short* __restrict__ vb) {
  const int lane = threadIdx.x & 31;
  const int wave = threadIdx.x >> 5;
  const int g = lane >> 4, c = lane & 15;
  const int m0 = blockIdx.x * 64 + (wave >> 1) * 32;
  const int n0 = blockIdx.y * 64 + (wave & 1) * 32;

  v8f acc[2][2] = {};
  for (int kk = 0; kk < DIM; kk += 32) {
    const unsigned short* a0p = xT + (size_t)(m0 + c) * DIM + kk;
    const unsigned short* a1p = a0p + 16 * DIM;
    const unsigned short* b0p = wq + (size_t)(n0 + c) * DIM + kk;
    const unsigned short* b1p = b0p + 16 * DIM;
    v16bf a0 = pack16(a0p + 8 * g, a0p + 16 + 8 * g);
    v16bf a1 = pack16(a1p + 8 * g, a1p + 16 + 8 * g);
    v16bf b0 = pack16(b0p + 8 * g, b0p + 16 + 8 * g);
    v16bf b1 = pack16(b1p + 8 * g, b1p + 16 + 8 * g);
    acc[0][0] = wmma_bf16(a0, b0, acc[0][0]);
    acc[0][1] = wmma_bf16(a0, b1, acc[0][1]);
    acc[1][0] = wmma_bf16(a1, b0, acc[1][0]);
    acc[1][1] = wmma_bf16(a1, b1, acc[1][1]);
  }
#pragma unroll
  for (int sm = 0; sm < 2; ++sm)
#pragma unroll
    for (int sn = 0; sn < 2; ++sn)
#pragma unroll
      for (int r = 0; r < 8; ++r) {
        int m = m0 + sm * 16 + r + 8 * g;        // token
        int n = n0 + sn * 16 + c;                // qkv column
        float vv = acc[sm][sn][r] + bqkv[n];
        int which = n >> 8, cc = n & 255;
        int head = cc >> 5, hd = cc & 31;
        int b = m >> 12, pix = m & 4095;
        size_t dst = ((size_t)((b * HEADS + head) * PIX + pix)) * HD + hd;
        if (which == 0)      qb[dst] = f2bf(vv * QSCALE);
        else if (which == 1) kb[dst] = f2bf(vv);
        else                 vb[dst] = f2bf(vv);
      }
}

// ---------- 3) fused neighborhood attention: 1 wave per (b, head, row, j-tile)
__global__ void natten_kernel(const unsigned short* __restrict__ qb,
                              const unsigned short* __restrict__ kb,
                              const unsigned short* __restrict__ vb,
                              const float* __restrict__ rpb,
                              unsigned short* __restrict__ attn) {
  __shared__ float          sscore[16 * NKEY];
  __shared__ unsigned short sprob[16 * NKEY];
  __shared__ int            keyoff[NKEY];

  const int lane = threadIdx.x;
  const int g = lane >> 4, c = lane & 15;
  const int b = blockIdx.z, head = blockIdx.y;
  const int i = blockIdx.x >> 2;
  const int j0 = (blockIdx.x & 3) * 16;

  int ni = i - 3;  ni  = ni  < 0 ? 0 : (ni  > 57 ? 57 : ni);
  int njm = j0 - 3; njm = njm < 0 ? 0 : (njm > 57 ? 57 : njm);
  const size_t bh = (size_t)(b * HEADS + head) * PIX;

  // key pixel offsets for the 7 x 22 (padded to 160) neighborhood union
  for (int key = lane; key < NKEY; key += 32) {
    int di = key / NCOL; if (di > 6) di = 6;
    int dc = key - di * NCOL;
    int col = njm + dc; if (col > 63) col = 63;
    keyoff[key] = (ni + di) * WW + col;
  }
  for (int t = lane; t < 16 * NKEY; t += 32) sprob[t] = 0;  // zero-padded probs
  __syncthreads();

  // A operand: 16 queries x hd=32
  const unsigned short* qrow = qb + (bh + (size_t)i * WW + (j0 + c)) * HD;
  v16bf aq = pack16(qrow + 8 * g, qrow + 16 + 8 * g);

  // scores = Q x K^T  (10 WMMAs -> LDS)
  for (int t = 0; t < 10; ++t) {
    const unsigned short* krow = kb + (bh + keyoff[t * 16 + c]) * HD;
    v16bf bk = pack16(krow + 8 * g, krow + 16 + 8 * g);
    v8f d = {};
    d = wmma_bf16(aq, bk, d);
#pragma unroll
    for (int r = 0; r < 8; ++r)
      sscore[(r + 8 * g) * NKEY + t * 16 + c] = d[r];
  }
  __syncthreads();

  // per-query softmax over its own 49 in-window keys (+rpb bias), f32
  if (lane < 16) {
    int j = j0 + lane;
    int njq = j - 3; njq = njq < 0 ? 0 : (njq > 57 ? 57 : njq);
    int off = njq - njm;
    int pi = ni - i + 6, pj = njq - j + 6;
    const float* rb = rpb + head * 169;
    float vals[49];
    float mx = -1e30f;
#pragma unroll
    for (int di = 0; di < 7; ++di)
#pragma unroll
      for (int dj = 0; dj < 7; ++dj) {
        float s = sscore[lane * NKEY + di * NCOL + off + dj] +
                  rb[(pi + di) * 13 + (pj + dj)];
        vals[di * 7 + dj] = s;
        mx = s > mx ? s : mx;
      }
    float sum = 0.f;
#pragma unroll
    for (int n = 0; n < 49; ++n) {
      float e = __expf(vals[n] - mx);
      vals[n] = e; sum += e;
    }
    float inv = 1.f / sum;
#pragma unroll
    for (int di = 0; di < 7; ++di)
#pragma unroll
      for (int dj = 0; dj < 7; ++dj)
        sprob[lane * NKEY + di * NCOL + off + dj] = f2bf(vals[di * 7 + dj] * inv);
  }
  __syncthreads();

  // out = P(16x160) x V(160x32): 5 K-chunks x 2 N-tiles = 10 WMMAs
  v8f o[2] = {};
  for (int chunk = 0; chunk < 5; ++chunk) {
    v16bf ap = pack16(&sprob[c * NKEY + chunk * 32 + 8 * g],
                      &sprob[c * NKEY + chunk * 32 + 16 + 8 * g]);
#pragma unroll
    for (int t2 = 0; t2 < 2; ++t2) {
      v16u bvv;
#pragma unroll
      for (int e = 0; e < 16; ++e) {
        int kd = chunk * 32 + (e < 8 ? 8 * g + e : 16 + 8 * g + (e - 8));
        bvv[e] = vb[(bh + keyoff[kd]) * HD + t2 * 16 + c];
      }
      v16bf bv = __builtin_bit_cast(v16bf, bvv);
      o[t2] = wmma_bf16(ap, bv, o[t2]);
    }
  }
#pragma unroll
  for (int t2 = 0; t2 < 2; ++t2)
#pragma unroll
    for (int r = 0; r < 8; ++r) {
      int m = r + 8 * g;
      size_t token = (size_t)b * PIX + (size_t)i * WW + j0 + m;
      attn[token * DIM + head * HD + t2 * 16 + c] = f2bf(o[t2][r]);
    }
}

// ---------- 4) proj GEMM: [8192x256]x[256x256]^T -> f32 channels-first out ---
__global__ void proj_gemm(const unsigned short* __restrict__ attn,
                          const unsigned short* __restrict__ wp,
                          const float* __restrict__ bproj,
                          float* __restrict__ out) {
  const int lane = threadIdx.x & 31;
  const int wave = threadIdx.x >> 5;
  const int g = lane >> 4, c = lane & 15;
  const int m0 = blockIdx.x * 64 + (wave >> 1) * 32;
  const int n0 = blockIdx.y * 64 + (wave & 1) * 32;

  v8f acc[2][2] = {};
  for (int kk = 0; kk < DIM; kk += 32) {
    const unsigned short* a0p = attn + (size_t)(m0 + c) * DIM + kk;
    const unsigned short* a1p = a0p + 16 * DIM;
    const unsigned short* b0p = wp + (size_t)(n0 + c) * DIM + kk;
    const unsigned short* b1p = b0p + 16 * DIM;
    v16bf a0 = pack16(a0p + 8 * g, a0p + 16 + 8 * g);
    v16bf a1 = pack16(a1p + 8 * g, a1p + 16 + 8 * g);
    v16bf b0 = pack16(b0p + 8 * g, b0p + 16 + 8 * g);
    v16bf b1 = pack16(b1p + 8 * g, b1p + 16 + 8 * g);
    acc[0][0] = wmma_bf16(a0, b0, acc[0][0]);
    acc[0][1] = wmma_bf16(a0, b1, acc[0][1]);
    acc[1][0] = wmma_bf16(a1, b0, acc[1][0]);
    acc[1][1] = wmma_bf16(a1, b1, acc[1][1]);
  }
#pragma unroll
  for (int sm = 0; sm < 2; ++sm)
#pragma unroll
    for (int sn = 0; sn < 2; ++sn)
#pragma unroll
      for (int r = 0; r < 8; ++r) {
        int m = m0 + sm * 16 + r + 8 * g;
        int n = n0 + sn * 16 + c;
        int b = m >> 12, pix = m & 4095;
        out[((size_t)(b * DIM + n)) * PIX + pix] = acc[sm][sn][r] + bproj[n];
      }
}

extern "C" void kernel_launch(void* const* d_in, const int* in_sizes, int n_in,
                              void* d_out, int out_size, void* d_ws, size_t ws_size,
                              hipStream_t stream) {
  (void)in_sizes; (void)n_in; (void)out_size; (void)ws_size;
  const float* x   = (const float*)d_in[0];
  const float* wq  = (const float*)d_in[1];
  const float* bq  = (const float*)d_in[2];
  const float* rpb = (const float*)d_in[3];
  const float* wp  = (const float*)d_in[4];
  const float* bp  = (const float*)d_in[5];
  float* out = (float*)d_out;

  char* ws = (char*)d_ws;
  size_t off = 0;
  auto alloc_u16 = [&](size_t elems) {
    unsigned short* p = (unsigned short*)(ws + off);
    off += ((elems * 2 + 255) & ~(size_t)255);
    return p;
  };
  unsigned short* xT  = alloc_u16((size_t)NTOK * DIM);       // x, token-major bf16
  unsigned short* wqb = alloc_u16((size_t)3 * DIM * DIM);    // w_qkv bf16
  unsigned short* wpb = alloc_u16((size_t)DIM * DIM);        // w_proj bf16
  unsigned short* qb  = alloc_u16((size_t)NTOK * DIM);       // Q (pre-scaled)
  unsigned short* kb  = alloc_u16((size_t)NTOK * DIM);       // K
  unsigned short* vb  = alloc_u16((size_t)NTOK * DIM);       // V
  unsigned short* at  = alloc_u16((size_t)NTOK * DIM);       // attention output

  const int prep_total = NTOK * DIM + 3 * DIM * DIM + DIM * DIM;  // 2359296
  prep_kernel<<<dim3((prep_total + 255) / 256), 256, 0, stream>>>(x, wq, wp, xT, wqb, wpb);
  qkv_gemm<<<dim3(NTOK / 64, (3 * DIM) / 64), 128, 0, stream>>>(xT, wqb, bq, qb, kb, vb);
  natten_kernel<<<dim3((HH * WW) / 16, HEADS, BATCH), 32, 0, stream>>>(qb, kb, vb, rpb, at);
  proj_gemm<<<dim3(NTOK / 64, DIM / 64), 128, 0, stream>>>(at, wpb, bp, out);
}
